// Qwen3Attention_22419729285772
// MI455X (gfx1250) — matrix operand order repaired and checked by reading, not compiled, not run on hardware
//
#include <hip/hip_runtime.h>
#include <hip/hip_bf16.h>
#include <stdint.h>

// ---------------------------------------------------------------------------
// Types
// ---------------------------------------------------------------------------
typedef __bf16 bf16;
typedef __bf16 v16bf __attribute__((ext_vector_type(16)));
typedef __bf16 v8bf  __attribute__((ext_vector_type(8)));
typedef __bf16 v4bf  __attribute__((ext_vector_type(4)));
typedef float  v8f   __attribute__((ext_vector_type(8)));
typedef float  v4f   __attribute__((ext_vector_type(4)));
// Exact element type expected by the async-to-LDS builtin (from the hipcc
// diagnostic: "__vector_size__(4 * sizeof(int)) int" pointers, AS1 / AS3).
typedef int v4i_b128 __attribute__((vector_size(16)));

#define HIDDEN 4096
#define NH     32
#define NKV    8
#define HD     128
#define SEQ    2048
#define BATCH  2
#define TOK    (BATCH * SEQ)       // 4096
#define QSZ    (NH * HD)           // 4096
#define KVSZ   (NKV * HD)          // 1024
#define QKV_N  (QSZ + 2 * KVSZ)    // 6144

#define AS1 __attribute__((address_space(1)))
#define AS3 __attribute__((address_space(3)))

#ifndef __has_builtin
#define __has_builtin(x) 0
#endif

// CDNA5 async global->LDS copy path (ASYNCcnt-tracked, bypasses VGPRs).
// Guarded: falls back to a VGPR round-trip copy if the toolchain lacks the
// builtins (host pass always takes the fallback branch, which is fine).
#if __has_builtin(__builtin_amdgcn_global_load_async_to_lds_b128)
#define HAVE_ASYNC_LDS 1
#else
#define HAVE_ASYNC_LDS 0
#endif

// ---------------------------------------------------------------------------
// Helpers
// ---------------------------------------------------------------------------
__device__ __forceinline__ bf16 f2bf(float f) { return (bf16)f; }  // native v_cvt

__device__ __forceinline__ v16bf cat8(v8bf a, v8bf b) {
  return __builtin_shufflevector(a, b, 0, 1, 2, 3, 4, 5, 6, 7,
                                       8, 9, 10, 11, 12, 13, 14, 15);
}

// A-matrix fragment (16-bit, 16x32): per-lane row, two 8-element k-chunks at
// k0 and k0+16 (k0 = laneHi*8 within the 32-wide K slab).
__device__ __forceinline__ v16bf frag_a16(const bf16* row, int k0) {
  v8bf lo = *reinterpret_cast<const v8bf*>(row + k0);
  v8bf hi = *reinterpret_cast<const v8bf*>(row + k0 + 16);
  return cat8(lo, hi);
}

// B-matrix fragment (16-bit, 32x16): per-lane column, two 8-element k-chunks at
// k0 and k0+16 (k0 = laneHi*8) (LDS holds B transposed: [n][k]).
__device__ __forceinline__ v16bf frag_b16(const bf16* row, int k0) {
  v8bf lo = *reinterpret_cast<const v8bf*>(row + k0);
  v8bf hi = *reinterpret_cast<const v8bf*>(row + k0 + 16);
  return cat8(lo, hi);
}

__device__ __forceinline__ v8f wmma_bf16(v16bf a, v16bf b, v8f c) {
  return __builtin_amdgcn_wmma_f32_16x16x32_bf16(
      /*neg_a=*/false, a, /*neg_b=*/false, b,
      /*c_mod=*/(short)0, c, /*reuse_a=*/false, /*reuse_b=*/false);
}

// 16-byte global -> LDS copy (8 bf16), async where available.
__device__ __forceinline__ void copy16_g2l(const bf16* g, bf16* l) {
#if HAVE_ASYNC_LDS
  __builtin_amdgcn_global_load_async_to_lds_b128(
      (AS1 v4i_b128*)g, (AS3 v4i_b128*)l, 0, 0);
#else
  *reinterpret_cast<v8bf*>(l) = *reinterpret_cast<const v8bf*>(g);
#endif
}

__device__ __forceinline__ void wait_async_copies() {
#if HAVE_ASYNC_LDS
#if __has_builtin(__builtin_amdgcn_s_wait_asynccnt)
  __builtin_amdgcn_s_wait_asynccnt(0);
#else
  asm volatile("s_wait_asynccnt 0" ::: "memory");
#endif
#endif
}

// Stage 8 A-elements into LDS: fp32 source converts via packed v_cvt;
// bf16 source goes through the async-to-LDS path.
__device__ __forceinline__ void stageA8(const float* g, bf16* l) {
  v4f f0 = *reinterpret_cast<const v4f*>(g);
  v4f f1 = *reinterpret_cast<const v4f*>(g + 4);
  *reinterpret_cast<v4bf*>(l)     = __builtin_convertvector(f0, v4bf);
  *reinterpret_cast<v4bf*>(l + 4) = __builtin_convertvector(f1, v4bf);
}
__device__ __forceinline__ void stageA8(const bf16* g, bf16* l) {
  copy16_g2l(g, l);
}

// ---------------------------------------------------------------------------
// Kernel 1/4: tiled bf16-WMMA GEMM  C[M,N](f32) = A[M,K] * B[K,N]
// A is fp32 or bf16 (fp32 converted during LDS staging; bf16 async-copied),
// B is fp32.  BM=BN=128, BK=32, 256 threads = 8 waves, wave -> 32x64 patch.
// ---------------------------------------------------------------------------
template <typename TA>
__global__ __launch_bounds__(256, 2)
void gemm_bf16_wmma(const TA* __restrict__ A, const float* __restrict__ B,
                    float* __restrict__ C, int M, int N, int K) {
  __shared__ __align__(16) bf16 la[128][32];   // [m][k]
  __shared__ __align__(16) bf16 lb[128][32];   // [n][k]  (B transposed)

  const int tid    = threadIdx.x;
  const int lane   = tid & 31;
  const int wave   = tid >> 5;
  const int laneHi = lane >> 4;
  const int lane16 = lane & 15;
  const int wm     = wave & 3;   // 4 waves along M (4*32 = 128)
  const int wn     = wave >> 2;  // 2 waves along N (2*64 = 128)
  const int mBase  = blockIdx.y * 128;
  const int nBase  = blockIdx.x * 128;

  v8f acc[2][4];
#pragma unroll
  for (int i = 0; i < 2; ++i)
#pragma unroll
    for (int j = 0; j < 4; ++j)
#pragma unroll
      for (int e = 0; e < 8; ++e) acc[i][j][e] = 0.0f;

  const int ktiles = K >> 5;
  for (int kt = 0; kt < ktiles; ++kt) {
    const int k0g = kt << 5;

    if (kt + 1 < ktiles) {  // L2 prefetch of next K-slab (global_prefetch_b8)
      __builtin_prefetch(&A[(size_t)(mBase + (tid >> 1)) * K + k0g + 32], 0, 1);
      __builtin_prefetch(&B[(size_t)(k0g + 32 + (tid >> 3)) * N + nBase], 0, 1);
    }

    // Stage A tile 128x32 (two 8-wide chunks per thread)
#pragma unroll
    for (int i = 0; i < 2; ++i) {
      const int c  = tid + 256 * i;      // 0..511 chunks of 8
      const int r  = c >> 2;             // 4 chunks per 32-wide row
      const int d0 = (c & 3) << 3;
      stageA8(&A[(size_t)(mBase + r) * K + k0g + d0], &la[r][d0]);
    }
    // Stage B tile 32x128 transposed into lb[n][k] (fp32 -> bf16 convert)
#pragma unroll
    for (int i = 0; i < 4; ++i) {
      const int q  = tid + 256 * i;
      const int kk = q >> 5;             // 32 quads per 128-wide row
      const int n4 = (q & 31) << 2;
      const v4f f =
          *reinterpret_cast<const v4f*>(&B[(size_t)(k0g + kk) * N + nBase + n4]);
      const v4bf b4 = __builtin_convertvector(f, v4bf);
      lb[n4 + 0][kk] = b4[0];
      lb[n4 + 1][kk] = b4[1];
      lb[n4 + 2][kk] = b4[2];
      lb[n4 + 3][kk] = b4[3];
    }
    wait_async_copies();
    __syncthreads();

    v16bf af[2], bfg[4];
#pragma unroll
    for (int tm = 0; tm < 2; ++tm)
      af[tm] = frag_a16(&la[wm * 32 + tm * 16 + lane16][0], laneHi * 8);
#pragma unroll
    for (int tn = 0; tn < 4; ++tn)
      bfg[tn] = frag_b16(&lb[wn * 64 + tn * 16 + lane16][0], laneHi * 8);

#pragma unroll
    for (int tm = 0; tm < 2; ++tm)
#pragma unroll
      for (int tn = 0; tn < 4; ++tn)
        acc[tm][tn] = wmma_bf16(af[tm], bfg[tn], acc[tm][tn]);

    __syncthreads();
  }

  // Epilogue: C-layout = row (r + 8*laneHi), col (lane16) per 16x16 tile
#pragma unroll
  for (int tm = 0; tm < 2; ++tm) {
    const int row0 = mBase + wm * 32 + tm * 16 + laneHi * 8;
#pragma unroll
    for (int tn = 0; tn < 4; ++tn) {
      const int col = nBase + wn * 64 + tn * 16 + lane16;
#pragma unroll
      for (int r = 0; r < 8; ++r)
        C[(size_t)(row0 + r) * N + col] = acc[tm][tn][r];
    }
  }
}

// ---------------------------------------------------------------------------
// Kernel 2/4: per-head RMSNorm + RoPE + repack to bf16 [b][h][s][d]
// grid = (TOK, 48): u<32 -> q head, 32..39 -> k head, 40..47 -> v head
// ---------------------------------------------------------------------------
__global__ __launch_bounds__(128)
void norm_rope_pack(const float* __restrict__ qkv, const int* __restrict__ pos,
                    const float* __restrict__ qw, const float* __restrict__ kw,
                    bf16* __restrict__ qo, bf16* __restrict__ ko,
                    bf16* __restrict__ vo) {
  const int t = blockIdx.x;
  const int u = blockIdx.y;
  const int d = threadIdx.x;
  const int b = t >> 11;
  const int s = t & (SEQ - 1);
  const size_t rowbase = (size_t)t * QKV_N;

  if (u >= NH + NKV) {  // V: plain bf16 convert + repack
    const int vh = u - (NH + NKV);
    const float x = qkv[rowbase + QSZ + KVSZ + vh * HD + d];
    vo[(((size_t)(b * NKV + vh)) * SEQ + s) * HD + d] = f2bf(x);
    return;
  }
  const bool isq = (u < NH);
  const int hh = isq ? u : (u - NH);
  const size_t hbase = rowbase + (isq ? hh * HD : QSZ + hh * HD);
  const float* w = isq ? qw : kw;

  const float x = qkv[hbase + d];
  float ss = x * x;
#pragma unroll
  for (int off = 1; off < 32; off <<= 1) ss += __shfl_xor(ss, off, 32);
  __shared__ float red[4];
  if ((threadIdx.x & 31) == 0) red[threadIdx.x >> 5] = ss;
  __syncthreads();
  const float tot = red[0] + red[1] + red[2] + red[3];
  const float rms = rsqrtf(tot * (1.0f / HD) + 1e-6f);

  const int   j    = d & 63;
  const float invf = __powf(1.0e6f, -(float)j * (1.0f / 64.0f));
  const float fr   = (float)pos[t] * invf;
  const float c  = __cosf(fr);
  const float sn = __sinf(fr);
  const float self = x * rms * w[d];
  const int   od   = (d < 64) ? (d + 64) : (d - 64);
  const float other = qkv[hbase + od] * rms * w[od];
  const float outv = (d < 64) ? (self * c - other * sn)
                              : (self * c + other * sn);

  if (isq) qo[(((size_t)(b * NH + hh)) * SEQ + s) * HD + d] = f2bf(outv);
  else     ko[(((size_t)(b * NKV + hh)) * SEQ + s) * HD + d] = f2bf(outv);
}

// ---------------------------------------------------------------------------
// Kernel 3/4: causal flash attention, bf16 WMMA, online softmax.
// One WG per (128-query tile, head, batch). 8 waves, 16 query rows each.
// Key blocks of 64. Q/K staged via async-to-LDS; V staged transposed [d][key]
// through VGPRs; P re-shaped via per-wave LDS buffer.
// ---------------------------------------------------------------------------
__global__ __launch_bounds__(256, 1)
void flash_attn_wmma(const bf16* __restrict__ Q, const bf16* __restrict__ K,
                     const bf16* __restrict__ V, bf16* __restrict__ O) {
  const int qt  = blockIdx.x;
  const int h   = blockIdx.y;
  const int b   = blockIdx.z;
  const int kvh = h >> 2;  // GQA: 32 q-heads over 8 kv-heads

  const int tid    = threadIdx.x;
  const int lane   = tid & 31;
  const int wave   = tid >> 5;
  const int laneHi = lane >> 4;
  const int lane16 = lane & 15;
  const int strip  = wave * 16;
  const int qbase  = qt * 128;

  const bf16* Qp = Q + ((size_t)(b * NH + h)) * SEQ * HD;
  const bf16* Kp = K + ((size_t)(b * NKV + kvh)) * SEQ * HD;
  const bf16* Vp = V + ((size_t)(b * NKV + kvh)) * SEQ * HD;

  union SMem {
    bf16 q[128][HD];                                 // 32 KB (Q staging phase)
    struct { bf16 k[64][HD]; bf16 vt[HD][64]; } kv;  // 16 KB + 16 KB
  };
  __shared__ __align__(16) SMem sm;
  __shared__ __align__(16) bf16 pbuf[8][16][64];     // per-wave P buffer, 16 KB

  // ---- Stage Q tile (async) and capture A-fragments in registers ----
#pragma unroll
  for (int i = 0; i < 8; ++i) {
    const int c  = tid + 256 * i;
    const int r  = c >> 4;
    const int d0 = (c & 15) << 3;
    copy16_g2l(&Qp[(size_t)(qbase + r) * HD + d0], &sm.q[r][d0]);
  }
  wait_async_copies();
  __syncthreads();
  v16bf qf[4];
#pragma unroll
  for (int kk = 0; kk < 4; ++kk)
    qf[kk] = frag_a16(&sm.q[strip + lane16][0], kk * 32 + laneHi * 8);
  __syncthreads();

  float m_r[8], l_r[8];
  v8f o_acc[8];
#pragma unroll
  for (int r = 0; r < 8; ++r) { m_r[r] = -1e30f; l_r[r] = 0.0f; }
#pragma unroll
  for (int dt = 0; dt < 8; ++dt)
#pragma unroll
    for (int e = 0; e < 8; ++e) o_acc[dt][e] = 0.0f;

  const float scale   = 0.08838834764831845f;  // 1/sqrt(128)
  const int   nblocks = (qbase >> 6) + 2;      // causal key coverage

  for (int kb = 0; kb < nblocks; ++kb) {
    const int key0 = kb << 6;

    // ---- Stage K [key][d] (async) and V transposed [d][key] (VGPR) ----
#pragma unroll
    for (int i = 0; i < 4; ++i) {
      const int c  = tid + 256 * i;
      const int r  = c >> 4;
      const int d0 = (c & 15) << 3;
      copy16_g2l(&Kp[(size_t)(key0 + r) * HD + d0], &sm.kv.k[r][d0]);
      const v8bf vv8 =
          *reinterpret_cast<const v8bf*>(&Vp[(size_t)(key0 + r) * HD + d0]);
#pragma unroll
      for (int jj = 0; jj < 8; ++jj) sm.kv.vt[d0 + jj][r] = vv8[jj];
    }
    wait_async_copies();
    __syncthreads();

    // ---- Scores: S(16x64) = Qstrip(16x128) * K^T(128x64) ----
    v8f sacc[4];
#pragma unroll
    for (int nt = 0; nt < 4; ++nt)
#pragma unroll
      for (int e = 0; e < 8; ++e) sacc[nt][e] = 0.0f;

#pragma unroll
    for (int nt = 0; nt < 4; ++nt) {
#pragma unroll
      for (int kk = 0; kk < 4; ++kk) {
        const v16bf bf_ =
            frag_b16(&sm.kv.k[nt * 16 + lane16][0], kk * 32 + laneHi * 8);
        sacc[nt] = wmma_bf16(qf[kk], bf_, sacc[nt]);
      }
    }

    // ---- Online softmax in C-layout (row = r + 8*laneHi, col = lane16) ----
#pragma unroll
    for (int r = 0; r < 8; ++r) {
      const int qrow = qbase + strip + r + laneHi * 8;
      float mx = -1e30f;
#pragma unroll
      for (int nt = 0; nt < 4; ++nt) {
        const int key = key0 + nt * 16 + lane16;
        float s = sacc[nt][r] * scale;
        if (key > qrow) s = -1e30f;  // causal mask
        sacc[nt][r] = s;
        mx = fmaxf(mx, s);
      }
#pragma unroll
      for (int off = 1; off < 16; off <<= 1)
        mx = fmaxf(mx, __shfl_xor(mx, off, 32));
      const float mnew  = fmaxf(m_r[r], mx);
      const float alpha = __expf(m_r[r] - mnew);
      float rs = 0.0f;
#pragma unroll
      for (int nt = 0; nt < 4; ++nt) {
        const float p = __expf(sacc[nt][r] - mnew);
        sacc[nt][r] = p;
        rs += p;
      }
#pragma unroll
      for (int off = 1; off < 16; off <<= 1) rs += __shfl_xor(rs, off, 32);
      l_r[r] = l_r[r] * alpha + rs;
      m_r[r] = mnew;
#pragma unroll
      for (int dt = 0; dt < 8; ++dt) o_acc[dt][r] *= alpha;
    }

    // ---- P -> per-wave LDS, re-read as A-fragments ----
#pragma unroll
    for (int nt = 0; nt < 4; ++nt)
#pragma unroll
      for (int r = 0; r < 8; ++r)
        pbuf[wave][r + laneHi * 8][nt * 16 + lane16] = f2bf(sacc[nt][r]);
    asm volatile("s_wait_dscnt 0" ::: "memory");  // own-wave LDS RAW fence

    // ---- O(16x128) += P(16x64) * V(64x128) ----
#pragma unroll
    for (int ks = 0; ks < 2; ++ks) {
      const v16bf pf = frag_a16(&pbuf[wave][lane16][0], ks * 32 + laneHi * 8);
#pragma unroll
      for (int dt = 0; dt < 8; ++dt) {
        const v16bf vf =
            frag_b16(&sm.kv.vt[dt * 16 + lane16][0], ks * 32 + laneHi * 8);
        o_acc[dt] = wmma_bf16(pf, vf, o_acc[dt]);
      }
    }
    __syncthreads();  // before next block overwrites K/V staging
  }

  // ---- Normalize and store attn output bf16 [b][s][h][d] ----
#pragma unroll
  for (int dt = 0; dt < 8; ++dt) {
    const int d = dt * 16 + lane16;
#pragma unroll
    for (int r = 0; r < 8; ++r) {
      const int srow = qbase + strip + r + laneHi * 8;
      const float val = o_acc[dt][r] / l_r[r];
      O[((size_t)(b * SEQ + srow)) * QSZ + h * HD + d] = f2bf(val);
    }
  }
}

// ---------------------------------------------------------------------------
// Launch
// ---------------------------------------------------------------------------
extern "C" void kernel_launch(void* const* d_in, const int* in_sizes, int n_in,
                              void* d_out, int out_size, void* d_ws,
                              size_t ws_size, hipStream_t stream) {
  const int*   positions = (const int*)d_in[0];
  const float* hidden    = (const float*)d_in[1];
  const float* w_qkv     = (const float*)d_in[2];
  const float* w_o       = (const float*)d_in[3];
  const float* q_norm_w  = (const float*)d_in[4];
  const float* k_norm_w  = (const float*)d_in[5];
  float*       out       = (float*)d_out;

  char* ws = (char*)d_ws;
  size_t off = 0;
  float* qkv  = (float*)(ws + off); off += (size_t)TOK * QKV_N * sizeof(float);
  bf16*  qb   = (bf16*)(ws + off);  off += (size_t)TOK * QSZ * sizeof(bf16);
  bf16*  kb   = (bf16*)(ws + off);  off += (size_t)TOK * KVSZ * sizeof(bf16);
  bf16*  vb   = (bf16*)(ws + off);  off += (size_t)TOK * KVSZ * sizeof(bf16);
  bf16*  attn = (bf16*)(ws + off);  off += (size_t)TOK * QSZ * sizeof(bf16);

  // 1) QKV projection: [4096,4096] x [4096,6144] -> fp32 qkv
  gemm_bf16_wmma<float><<<dim3(QKV_N / 128, TOK / 128), 256, 0, stream>>>(
      hidden, w_qkv, qkv, TOK, QKV_N, HIDDEN);

  // 2) RMSNorm + RoPE + bf16 repack
  norm_rope_pack<<<dim3(TOK, NH + 2 * NKV), 128, 0, stream>>>(
      qkv, positions, q_norm_w, k_norm_w, qb, kb, vb);

  // 3) Causal flash attention
  flash_attn_wmma<<<dim3(SEQ / 128, NH, BATCH), 256, 0, stream>>>(
      qb, kb, vb, attn);

  // 4) Output projection: [4096,4096](bf16) x [4096,4096] -> fp32 out
  gemm_bf16_wmma<bf16><<<dim3(HIDDEN / 128, TOK / 128), 256, 0, stream>>>(
      attn, w_o, out, TOK, HIDDEN, QSZ);
}
